// TernaryKVCache_46832323395754
// MI455X (gfx1250) — matrix operand order repaired and checked by reading, not compiled, not run on hardware
//
#include <hip/hip_runtime.h>

#define B_ 8
#define H_ 32
#define TQ_ 8
#define KV_ 4096
#define D_ 128
#define WAVES 8
#define STEP 32                    // kv columns per inner step
#define SLICE (KV_ / WAVES)        // 512 kv per wave
#define NSTEP (SLICE / STEP)       // 16 steps

typedef _Float16 v16h __attribute__((ext_vector_type(16)));
typedef float    v8f  __attribute__((ext_vector_type(8)));
typedef float    f4   __attribute__((ext_vector_type(4)));

union HPack { _Float16 h[2]; unsigned int u; };

__global__ __launch_bounds__(WAVES * 32, 1)
void ternary_attn_kernel(const float* __restrict__ qg,
                         const float* __restrict__ kg,
                         const float* __restrict__ vg,
                         const float* __restrict__ logt,
                         float* __restrict__ og)
{
    // per-wave V^T staging tile (f16), kv contiguous:  8 * 128 * 32 * 2B = 64 KB
    __shared__ __align__(32) _Float16 vt[WAVES][D_][STEP];
    __shared__ float obuf[TQ_ * D_];       // cross-wave output accumulator (4 KB)
    __shared__ float mbuf[WAVES][TQ_];
    __shared__ float lbuf[WAVES][TQ_];

    const int bh = blockIdx.x;
    const int h  = bh % H_;
    const float* qp = qg + (size_t)bh * TQ_ * D_;
    const float* kp = kg + (size_t)bh * KV_ * D_;
    const float* vp = vg + (size_t)bh * KV_ * D_;
    float*       op = og + (size_t)bh * TQ_ * D_;

    const int tid  = threadIdx.x;
    const int wid  = tid >> 5;
    const int lane = tid & 31;
    const int col  = lane & 15;   // q column (Sᵀ) / d column (AV) / kv-pair id (V stage)
    const int hi   = lane >> 4;   // lane half selects K-range of fragments

    const float xt     = logt[h];
    const float thresh = (xt > 20.0f) ? xt : log1pf(__expf(xt));
    const float sscale = thresh * 0.088388347648318447f;   // thresh / sqrt(D)

    for (int i = tid; i < TQ_ * D_; i += WAVES * 32) obuf[i] = 0.0f;
    __syncthreads();

    // ---- Q^T B-fragments (one per 32-wide d chunk), built once.
    // B 32x16 layout: lanes<16 hold K(d)=d0..d0+15, lanes>=16 hold next 16, pair-packed.
    v16h bQ[4];
#pragma unroll
    for (int dc = 0; dc < 4; ++dc) {
        const int d0 = dc * 32 + hi * 16;
        if (col < TQ_) {
            const f4* src = (const f4*)(qp + col * D_ + d0);
#pragma unroll
            for (int c4 = 0; c4 < 4; ++c4) {
                f4 fv = src[c4];
#pragma unroll
                for (int i = 0; i < 4; ++i) bQ[dc][c4 * 4 + i] = (_Float16)fv[i];
            }
        } else {
#pragma unroll
            for (int i = 0; i < 16; ++i) bQ[dc][i] = (_Float16)0.0f;
        }
    }

    v8f acc[8];
    const v8f zero8 = {0.f, 0.f, 0.f, 0.f, 0.f, 0.f, 0.f, 0.f};
#pragma unroll
    for (int t = 0; t < 8; ++t) acc[t] = zero8;
    float m = -1e30f, l = 0.0f;

    const int kv0 = wid * SLICE;

    for (int s = 0; s < NSTEP; ++s) {
        const int kvb = kv0 + s * STEP;

        // ---- stage V chunk into LDS transposed as f16: vt[d][kv], kv pair-packed b32 stores
        {
            const float* r0 = vp + (size_t)(kvb + 2 * col)     * D_ + hi * 64;
            const float* r1 = vp + (size_t)(kvb + 2 * col + 1) * D_ + hi * 64;
#pragma unroll
            for (int dch = 0; dch < 64; dch += 4) {
                f4 a = *(const f4*)(r0 + dch);
                f4 b = *(const f4*)(r1 + dch);
#pragma unroll
                for (int i = 0; i < 4; ++i) {
                    HPack pk;
                    pk.h[0] = (_Float16)a[i];
                    pk.h[1] = (_Float16)b[i];
                    *(unsigned int*)&vt[wid][hi * 64 + dch + i][2 * col] = pk.u;
                }
            }
        }

        // prefetch next step's K chunk (gfx1250 global_prefetch_b8)
        if (s + 1 < NSTEP)
            __builtin_prefetch(kp + (size_t)(kvb + STEP) * D_ + lane * 128, 0, 0);

        // ---- S^T = ternary(K) x Q^T : two 16(kv) x 16(q) tiles, contraction d in 4 chunks
        v8f st0 = zero8, st1 = zero8;
#pragma unroll
        for (int t = 0; t < 2; ++t) {
#pragma unroll
            for (int dc = 0; dc < 4; ++dc) {
                // A 16x32 f16 layout: lanes<16 row=col, d {dc*32+0..15};
                // lanes>=16 same row set, d {+16..31}
                const float* kr = kp + (size_t)(kvb + t * 16 + col) * D_;
                const int s0 = dc * 32 + hi * 16;
                const int s1 = dc * 32 + hi * 16 + 8;
                f4 k0 = *(const f4*)(kr + s0);
                f4 k1 = *(const f4*)(kr + s0 + 4);
                f4 k2 = *(const f4*)(kr + s1);
                f4 k3 = *(const f4*)(kr + s1 + 4);
                v16h aK;
#pragma unroll
                for (int i = 0; i < 4; ++i) {
                    aK[i]      = (_Float16)((k0[i] > thresh) ? 1.0f : ((k0[i] < -thresh) ? -1.0f : 0.0f));
                    aK[4 + i]  = (_Float16)((k1[i] > thresh) ? 1.0f : ((k1[i] < -thresh) ? -1.0f : 0.0f));
                    aK[8 + i]  = (_Float16)((k2[i] > thresh) ? 1.0f : ((k2[i] < -thresh) ? -1.0f : 0.0f));
                    aK[12 + i] = (_Float16)((k3[i] > thresh) ? 1.0f : ((k3[i] < -thresh) ? -1.0f : 0.0f));
                }
                if (t == 0)
                    st0 = __builtin_amdgcn_wmma_f32_16x16x32_f16(false, aK, false, bQ[dc],
                                                                 (short)0, st0, false, false);
                else
                    st1 = __builtin_amdgcn_wmma_f32_16x16x32_f16(false, aK, false, bQ[dc],
                                                                 (short)0, st1, false, false);
            }
        }

        // ---- online softmax; row = q = col (replicated in lanes col and col+16)
        float sv[16];
#pragma unroll
        for (int j = 0; j < 8; ++j) {
            sv[j]     = st0[j] * sscale;
            sv[8 + j] = st1[j] * sscale;
        }

        float mloc = -1e30f;
#pragma unroll
        for (int i = 0; i < 16; ++i) mloc = fmaxf(mloc, sv[i]);
        mloc = fmaxf(mloc, __shfl_xor(mloc, 16, 32));
        const float mnew = fmaxf(m, mloc);
        const float corr = __expf(m - mnew);
        m = mnew;

        float p[16];
        float ls = 0.0f;
#pragma unroll
        for (int i = 0; i < 16; ++i) { p[i] = __expf(sv[i] - m); ls += p[i]; }
        ls += __shfl_xor(ls, 16, 32);
        l = l * corr + ls;

        // per-row correction for accumulator rows j (+8*hi). Rows 8..15 (hi=1 half)
        // are padding, so a uniform-source shuffle (readlane broadcast) is enough.
        float cj[8];
#pragma unroll
        for (int j = 0; j < 8; ++j) cj[j] = __shfl(corr, j, 32);

        // S^T C-layout maps directly onto the A-fragment layout for P(16q x 32kv)
        v16h pA;
#pragma unroll
        for (int i = 0; i < 16; ++i) pA[i] = (_Float16)p[i];

        // ---- acc += P x V : 8 d-tiles, contraction kv = 32
#pragma unroll
        for (int dt = 0; dt < 8; ++dt) {
#pragma unroll
            for (int j = 0; j < 8; ++j) acc[dt][j] *= cj[j];
            const _Float16* vrow = &vt[wid][dt * 16 + col][0];
            v16h bV;
#pragma unroll
            for (int i = 0; i < 8; ++i) { bV[i] = vrow[hi * 8 + i]; bV[8 + i] = vrow[16 + hi * 8 + i]; }
            acc[dt] = __builtin_amdgcn_wmma_f32_16x16x32_f16(false, pA, false, bV,
                                                             (short)0, acc[dt], false, false);
        }
    }

    // ---- cross-wave combine via LDS
    if (hi == 0 && col < TQ_) { mbuf[wid][col] = m; lbuf[wid][col] = l; }
    __syncthreads();

    if (hi == 0) {  // lanes 0..15 hold the 8 real query rows (j = q), col = d within tile
#pragma unroll
        for (int j = 0; j < TQ_; ++j) {
            float Mq = -1e30f;
            for (int w = 0; w < WAVES; ++w) Mq = fmaxf(Mq, mbuf[w][j]);
            const float wgt = __expf(mbuf[wid][j] - Mq);
#pragma unroll
            for (int dt = 0; dt < 8; ++dt)
                atomicAdd(&obuf[j * D_ + dt * 16 + col], wgt * acc[dt][j]);
        }
    }
    __syncthreads();

    for (int i = tid; i < TQ_ * D_; i += WAVES * 32) {
        const int qq = i >> 7;  // i / D_
        float Mq = -1e30f;
        for (int w = 0; w < WAVES; ++w) Mq = fmaxf(Mq, mbuf[w][qq]);
        float Ls = 0.0f;
        for (int w = 0; w < WAVES; ++w) Ls += __expf(mbuf[w][qq] - Mq) * lbuf[w][qq];
        op[i] = obuf[i] / Ls;
    }
}

extern "C" void kernel_launch(void* const* d_in, const int* in_sizes, int n_in,
                              void* d_out, int out_size, void* d_ws, size_t ws_size,
                              hipStream_t stream) {
    const float* q  = (const float*)d_in[0];
    const float* k  = (const float*)d_in[1];
    const float* v  = (const float*)d_in[2];
    const float* lt = (const float*)d_in[3];
    float* out = (float*)d_out;
    (void)in_sizes; (void)n_in; (void)out_size; (void)d_ws; (void)ws_size;

    dim3 grid(B_ * H_);          // one workgroup per (b, h)
    dim3 block(WAVES * 32);      // 8 wave32s, each owns KV/8 = 512
    hipLaunchKernelGGL(ternary_attn_kernel, grid, block, 0, stream, q, k, v, lt, out);
}
